// Model_57982058496057
// MI455X (gfx1250) — matrix operand order repaired and checked by reading, not compiled, not run on hardware
//
#include <hip/hip_runtime.h>
#include <stdint.h>

#define N_TOK 4096
#define DIMK  2048
#define NEXP  8
#define FDIM  4096
#define CAP   4096       // max tokens per expert (worst case)
#define KT    32         // K step (bf16 wmma 16x16x32)
#define BM    128        // block rows (token slots)
#define BN    256        // block cols (features)
#define LDR   80         // LDS row stride bytes: 32 bf16 = 64B + 16B pad (conflict-free b128)
#define A_TILE (BM * LDR)          // 10240
#define B_TILE (BN * LDR)          // 20480
#define SM_TOK 0
#define SM_W   512
#define SM_A   1024
#define SM_B   (SM_A + 2 * A_TILE)             // 21504
#define SM_TOTAL (SM_B + 2 * B_TILE)           // 62464

typedef unsigned int uint32;
typedef __attribute__((ext_vector_type(16))) __bf16 v16bf;
typedef __attribute__((ext_vector_type(8)))  float  v8f;

union FragAB { uint4 u[2]; v16bf v; };

__device__ __forceinline__ unsigned short f2bf(float f) {
  uint32 u = __float_as_uint(f);
  u += 0x7FFFu + ((u >> 16) & 1u);   // round-to-nearest-even
  return (unsigned short)(u >> 16);
}

__device__ __forceinline__ void async_b128(uint32 lds_off, const void* gptr) {
  asm volatile("global_load_async_to_lds_b128 %0, %1, off"
               :: "v"(lds_off), "v"((unsigned long long)(uintptr_t)gptr)
               : "memory");
}

// ---------------- zero output ----------------
__global__ void k_zero_f4(float4* p, int n4) {
  int i = blockIdx.x * blockDim.x + threadIdx.x;
  int stride = gridDim.x * blockDim.x;
  float4 z = make_float4(0.f, 0.f, 0.f, 0.f);
  for (; i < n4; i += stride) p[i] = z;
}

// ---------------- init routing arrays ----------------
__global__ void k_init_slots(int* tok, float* w, uint32* counts) {
  int i = blockIdx.x * blockDim.x + threadIdx.x;
  if (i < NEXP * CAP) { tok[i] = 0; w[i] = 0.f; }
  if (i < NEXP) counts[i] = 0u;
}

// ---------------- f32 -> bf16 ----------------
__global__ void k_cvt_bf16(const float4* __restrict__ src, ushort4* __restrict__ dst, int n4) {
  int i = blockIdx.x * blockDim.x + threadIdx.x;
  int stride = gridDim.x * blockDim.x;
  for (; i < n4; i += stride) {
    float4 v = src[i];
    ushort4 o;
    o.x = f2bf(v.x); o.y = f2bf(v.y); o.z = f2bf(v.z); o.w = f2bf(v.w);
    dst[i] = o;
  }
}

// ---------------- gating + top-2 routing: one wave32 per token ----------------
__global__ void k_route(const float* __restrict__ x, const float* __restrict__ gw,
                        int* __restrict__ tok, float* __restrict__ w,
                        uint32* __restrict__ counts) {
  int wave = threadIdx.x >> 5;
  int lane = threadIdx.x & 31;
  int token = blockIdx.x * 8 + wave;
  const float4* xr = (const float4*)(x + (size_t)token * DIMK);
  float acc[NEXP];
#pragma unroll
  for (int e = 0; e < NEXP; ++e) acc[e] = 0.f;
  for (int d = lane; d < DIMK / 4; d += 32) {
    float4 xv = xr[d];
#pragma unroll
    for (int e = 0; e < NEXP; ++e) {
      float4 g = ((const float4*)(gw + (size_t)e * DIMK))[d];
      acc[e] += xv.x * g.x + xv.y * g.y + xv.z * g.z + xv.w * g.w;
    }
  }
#pragma unroll
  for (int e = 0; e < NEXP; ++e) {
#pragma unroll
    for (int s = 16; s >= 1; s >>= 1) acc[e] += __shfl_xor(acc[e], s, 32);
  }
  if (lane == 0) {
    int b0 = 0;
#pragma unroll
    for (int e = 1; e < NEXP; ++e) if (acc[e] > acc[b0]) b0 = e;
    int b1 = (b0 == 0) ? 1 : 0;
#pragma unroll
    for (int e = 0; e < NEXP; ++e) if (e != b0 && acc[e] > acc[b1]) b1 = e;
    // softmax over experts then renorm of top-2 == softmax over {s0,s1}
    float e1 = __expf(acc[b1] - acc[b0]);
    float w0 = 1.f / (1.f + e1);
    float w1 = e1 * w0;
    uint32 p0 = atomicAdd(&counts[b0], 1u);
    tok[b0 * CAP + p0] = token; w[b0 * CAP + p0] = w0;
    uint32 p1 = atomicAdd(&counts[b1], 1u);
    tok[b1 * CAP + p1] = token; w[b1 * CAP + p1] = w1;
  }
}

// ------- grouped expert GEMM, 128x256 block tile, wave tile 64x64, bf16 WMMA -------
__global__ void __launch_bounds__(256)
k_moe_gemm(const unsigned short* __restrict__ xbf,
           const unsigned short* __restrict__ wbf,
           const int* __restrict__ tok,
           const float* __restrict__ wgt,
           const uint32* __restrict__ counts,
           float* __restrict__ out) {
  const int e  = blockIdx.z;
  const int m0 = blockIdx.y * BM;    // slot base within expert
  const int n0 = blockIdx.x * BN;    // feature base
  if ((uint32)m0 >= counts[e]) return;

  __shared__ __align__(16) char smem[SM_TOTAL];
  int*   sTok = (int*)(smem + SM_TOK);
  float* sW   = (float*)(smem + SM_W);

  const int tid = threadIdx.x;
  if (tid < BM) {
    sTok[tid] = tok[e * CAP + m0 + tid];
    sW[tid]   = wgt[e * CAP + m0 + tid];
  }
  __syncthreads();

  const uint32 smemBase = (uint32)(uintptr_t)smem;

  // global->LDS async assignment:
  //   A: BM*4 = 512 b128 chunks  -> 2 per thread
  //   B: BN*4 = 1024 b128 chunks -> 4 per thread
  const int ar0 = tid >> 2,         as0 = tid & 3;
  const int ar1 = (tid + 256) >> 2, as1 = (tid + 256) & 3;
  const unsigned short* aG0 = xbf + (size_t)sTok[ar0] * DIMK + as0 * 8;
  const unsigned short* aG1 = xbf + (size_t)sTok[ar1] * DIMK + as1 * 8;
  const uint32 aL0 = ar0 * LDR + as0 * 16;
  const uint32 aL1 = ar1 * LDR + as1 * 16;

  const unsigned short* bG[4];
  uint32 bL[4];
#pragma unroll
  for (int j = 0; j < 4; ++j) {
    int c = tid + j * 256;
    int br = c >> 2, bs = c & 3;
    bG[j] = wbf + ((size_t)e * FDIM + n0 + br) * DIMK + bs * 8;
    bL[j] = br * LDR + bs * 16;
  }

  // wave tiling: 8 waves as 2 (rows) x 4 (cols); each wave: 64x64 = 4x4 accs
  const int waveId = tid >> 5;
  const int lane   = tid & 31;
  const int wm = waveId >> 2;
  const int wn = waveId & 3;
  const int lmod  = lane & 15;
  const int lhalf = lane >> 4;

  uint32 aOff[4][2], bOff[4][2];
#pragma unroll
  for (int m = 0; m < 4; ++m) {
    int row = wm * 64 + m * 16 + lmod;               // A 16-bit layout (ISA 7.12.2)
    aOff[m][0] = row * LDR + lhalf * 16;             // K 0..7  / 8..15
    aOff[m][1] = row * LDR + 32 + lhalf * 16;        // K 16..23 / 24..31
  }
#pragma unroll
  for (int n = 0; n < 4; ++n) {
    int row = wn * 64 + n * 16 + lmod;               // B: same K order as A (K 0..7/8..15, 16..23/24..31)
    bOff[n][0] = row * LDR + lhalf * 16;
    bOff[n][1] = row * LDR + 32 + lhalf * 16;
  }

  v8f acc[4][4];
#pragma unroll
  for (int m = 0; m < 4; ++m)
#pragma unroll
    for (int n = 0; n < 4; ++n)
#pragma unroll
      for (int i = 0; i < 8; ++i) acc[m][n][i] = 0.f;

  auto issue = [&](int k0, int b) {
    uint32 ab = smemBase + SM_A + b * A_TILE;
    uint32 bb = smemBase + SM_B + b * B_TILE;
    async_b128(ab + aL0, aG0 + k0);
    async_b128(ab + aL1, aG1 + k0);
#pragma unroll
    for (int j = 0; j < 4; ++j) async_b128(bb + bL[j], bG[j] + k0);
  };

  auto compute = [&](int b) {
    const char* aB = smem + SM_A + b * A_TILE;
    const char* bB = smem + SM_B + b * B_TILE;
    FragAB A[4], B[4];
#pragma unroll
    for (int m = 0; m < 4; ++m) {
      A[m].u[0] = *(const uint4*)(aB + aOff[m][0]);
      A[m].u[1] = *(const uint4*)(aB + aOff[m][1]);
    }
#pragma unroll
    for (int n = 0; n < 4; ++n) {
      B[n].u[0] = *(const uint4*)(bB + bOff[n][0]);
      B[n].u[1] = *(const uint4*)(bB + bOff[n][1]);
    }
#pragma unroll
    for (int n = 0; n < 4; ++n)
#pragma unroll
      for (int m = 0; m < 4; ++m)
        acc[m][n] = __builtin_amdgcn_wmma_f32_16x16x32_bf16(
            false, A[m].v, false, B[n].v, (short)0, acc[m][n], false, false);
  };

  issue(0, 0);
  int buf = 0;
  // steady state: branch-free body, last K-step peeled
  for (int k0 = 0; k0 < DIMK - KT; k0 += KT) {
    issue(k0 + KT, buf ^ 1);
    asm volatile("s_wait_asynccnt 0x6" ::: "memory");   // current buffer's 6 done
    __syncthreads();
    compute(buf);
    __syncthreads();   // all waves done reading before next issue overwrites
    buf ^= 1;
  }
  asm volatile("s_wait_asynccnt 0x0" ::: "memory");
  __syncthreads();
  compute(buf);

  // epilogue: D 16x16 f32 layout: VGPR v -> M = v + 8*lhalf, N = lane%16
#pragma unroll
  for (int m = 0; m < 4; ++m) {
    int rbase = wm * 64 + m * 16;
#pragma unroll
    for (int n = 0; n < 4; ++n) {
      int col = n0 + wn * 64 + n * 16 + lmod;
#pragma unroll
      for (int v = 0; v < 8; ++v) {
        int r = rbase + v + lhalf * 8;
        float val = acc[m][n][v] * sW[r];               // padded rows: sW==0 -> adds 0
        float* dst = out + (size_t)sTok[r] * FDIM + col;
        __hip_atomic_fetch_add(dst, val, __ATOMIC_RELAXED, __HIP_MEMORY_SCOPE_AGENT);
      }
    }
  }
}

extern "C" void kernel_launch(void* const* d_in, const int* in_sizes, int n_in,
                              void* d_out, int out_size, void* d_ws, size_t ws_size,
                              hipStream_t stream) {
  const float* x  = (const float*)d_in[0];   // [N, DIM]
  const float* gw = (const float*)d_in[1];   // [E, DIM]
  const float* ew = (const float*)d_in[2];   // [E, F, DIM]
  float* out = (float*)d_out;                // [N, F]

  char* ws = (char*)d_ws;
  size_t off = 0;
  unsigned short* xbf = (unsigned short*)(ws + off); off += (size_t)N_TOK * DIMK * 2;
  unsigned short* wbf = (unsigned short*)(ws + off); off += (size_t)NEXP * FDIM * DIMK * 2;
  int*    tok    = (int*)(ws + off);    off += (size_t)NEXP * CAP * 4;
  float*  wgt    = (float*)(ws + off);  off += (size_t)NEXP * CAP * 4;
  uint32* counts = (uint32*)(ws + off); off += 64;

  k_zero_f4<<<4096, 256, 0, stream>>>((float4*)out, N_TOK * FDIM / 4);
  k_init_slots<<<(NEXP * CAP + 255) / 256, 256, 0, stream>>>(tok, wgt, counts);
  k_cvt_bf16<<<2048, 256, 0, stream>>>((const float4*)x, (ushort4*)xbf, N_TOK * DIMK / 4);
  k_cvt_bf16<<<8192, 256, 0, stream>>>((const float4*)ew, (ushort4*)wbf, NEXP * FDIM * DIMK / 4);
  k_route<<<N_TOK / 8, 256, 0, stream>>>(x, gw, tok, wgt, counts);
  k_moe_gemm<<<dim3(FDIM / BN, CAP / BM, NEXP), 256, 0, stream>>>(xbf, wbf, tok, wgt, counts, out);
}
